// PM_Model_24086176596518
// MI455X (gfx1250) — matrix operand order repaired and checked by reading, not compiled, not run on hardware
//
#include <hip/hip_runtime.h>
#include <math.h>

typedef __attribute__((ext_vector_type(16))) __bf16 v16bf;
typedef __attribute__((ext_vector_type(8)))  float  v8f;

static __device__ __forceinline__ float lrelu(float x) { return x > 0.f ? x : 0.01f * x; }

// ---------------------------------------------------------------------------
// Elementwise / setup kernels
// ---------------------------------------------------------------------------

__global__ void k_emb_init(float* __restrict__ emb, const float* __restrict__ feat,
                           long long total, long long slice) {
  long long i = (long long)blockIdx.x * blockDim.x + threadIdx.x;
  if (i >= total) return;
  emb[i] = feat[i % slice];
}

__global__ void k_rels_init(float* __restrict__ rels, const float* __restrict__ rel_emb) {
  int t = blockIdx.x * blockDim.x + threadIdx.x;
  if (t >= 512) return;
  int k = t >> 7, j = t & 127;
  int p = (k == 0) ? 3 : (k - 1);  // processing order (n,poi,s,d) = rel_emb[{3,0,1,2}]
  rels[t] = rel_emb[p * 128 + j];
}

__global__ void k_deg(float* __restrict__ deg, const int* __restrict__ edges,
                      long long E, int N) {
  long long i = (long long)blockIdx.x * blockDim.x + threadIdx.x;
  if (i >= 4 * E) return;
  int r = (int)(i / E);
  long long e = i % E;
  int p = (r == 0) ? 3 : (r - 1);
  int dst = edges[(long long)p * 2 * E + E + e];
  atomicAdd(&deg[(long long)r * N + dst], 1.0f);
}

__global__ void k_dinv(float* __restrict__ dinv, long long total) {
  long long i = (long long)blockIdx.x * blockDim.x + threadIdx.x;
  if (i >= total) return;
  dinv[i] = rsqrtf(dinv[i] + 1.0f);  // deg(dst) + self-loop
}

// Pack a [128,Nout] fp32 weight (optionally transposed [Nout,128], optionally
// row-scaled per batch by scale[b][K]) into the bf16 WMMA B-fragment layout:
// within batch: idx = ((nT*4 + kt)*32 + lane)*16 + j,
// K = kt*32 + ((lane>=16)?8:0) + (j<8 ? j : j+8), n = nT*16 + (lane&15).
__global__ void k_pack(__bf16* __restrict__ hi, __bf16* __restrict__ lo,
                       const float* __restrict__ W, const float* __restrict__ scale,
                       int transposed, int Nout, long long total) {
  long long t = (long long)blockIdx.x * blockDim.x + threadIdx.x;
  if (t >= total) return;
  long long per = (long long)Nout * 128;
  int b = (int)(t / per);
  long long rem = t % per;
  int j    = (int)(rem & 15);
  int lane = (int)((rem >> 4) & 31);
  int kt   = (int)((rem >> 9) & 3);
  int nt   = (int)(rem >> 11);
  int K = kt * 32 + ((lane & 16) ? 8 : 0) + (j < 8 ? j : j + 8);
  int n = nt * 16 + (lane & 15);
  float v = transposed ? W[(long long)n * 128 + K] : W[(long long)K * Nout + n];
  if (scale) v *= scale[b * 128 + K];
  __bf16 h = (__bf16)v;
  hi[t] = h;
  lo[t] = (__bf16)(v - (float)h);
}

// ---------------------------------------------------------------------------
// Batched bf16 split-precision WMMA GEMM: C[b] = A[amap[b]] (MxK=128) * B[b] + bias
// One wave computes a 16x64 strip (4 adjacent 16x16 N-tiles): the fp32 A panel
// is loaded and hi/lo-split once per K-slab and reused across 4 B tiles
// (48 WMMAs per wave, A-traffic / 4). K=128 as 4 slabs of 32; 3 WMMAs/slab
// (hi*hi, hi*lo, lo*hi) give near-fp32 accuracy.  Requires Nout % 64 == 0.
// ---------------------------------------------------------------------------

__global__ void k_gemm(const float* __restrict__ A, long long aStride,
                       int m0, int m1, int m2, int m3, int rowOff,
                       const __bf16* __restrict__ Bhi, const __bf16* __restrict__ Blo,
                       long long bStride,
                       const float* __restrict__ bias,
                       float* __restrict__ C, long long cStride,
                       int M, int Nout) {
  const int lane = threadIdx.x & 31;
  const int wave = blockIdx.x * (blockDim.x >> 5) + (threadIdx.x >> 5);
  const int stripsN = Nout >> 6;               // strips of 4 N-tiles
  const int tilesM  = (M + 15) >> 4;
  if (wave >= tilesM * stripsN) return;
  const int mT = wave / stripsN, nS = wave % stripsN;
  const int b = blockIdx.y;
  const int am = (b == 0) ? m0 : (b == 1) ? m1 : (b == 2) ? m2 : m3;
  const float*  Ab = A + (long long)am * aStride + (long long)rowOff * 128;
  const __bf16* bh = Bhi + (long long)b * bStride;
  const __bf16* bl = Blo + (long long)b * bStride;
  // A fragment: lane holds row M = mT*16 + lane%16, K halves at base=(lane>=16)?8:0
  const int rloc  = min(mT * 16 + (lane & 15), M - 1);
  const int abase = (lane & 16) ? 8 : 0;
  v8f acc0 = {}, acc1 = {}, acc2 = {}, acc3 = {};
#pragma unroll
  for (int kt = 0; kt < 4; ++kt) {
    const float* ap = Ab + (long long)rloc * 128 + kt * 32 + abase;
    float4 x0 = *(const float4*)(ap);
    float4 x1 = *(const float4*)(ap + 4);
    float4 x2 = *(const float4*)(ap + 16);
    float4 x3 = *(const float4*)(ap + 20);
    float xs[16] = {x0.x, x0.y, x0.z, x0.w, x1.x, x1.y, x1.z, x1.w,
                    x2.x, x2.y, x2.z, x2.w, x3.x, x3.y, x3.z, x3.w};
    v16bf ahi, alo;
#pragma unroll
    for (int j = 0; j < 16; ++j) {
      __bf16 hv = (__bf16)xs[j];
      ahi[j] = hv;
      alo[j] = (__bf16)(xs[j] - (float)hv);
    }
#pragma unroll
    for (int t = 0; t < 4; ++t) {
      const long long fb = (((long long)((nS * 4 + t) * 4 + kt)) * 32 + lane) * 16;
      v16bf bhi = *(const v16bf*)(bh + fb);
      v16bf blo = *(const v16bf*)(bl + fb);
      v8f acc = (t == 0) ? acc0 : (t == 1) ? acc1 : (t == 2) ? acc2 : acc3;
      acc = __builtin_amdgcn_wmma_f32_16x16x32_bf16(false, ahi, false, bhi, (short)0, acc, false, false);
      acc = __builtin_amdgcn_wmma_f32_16x16x32_bf16(false, ahi, false, blo, (short)0, acc, false, false);
      acc = __builtin_amdgcn_wmma_f32_16x16x32_bf16(false, alo, false, bhi, (short)0, acc, false, false);
      if (t == 0) acc0 = acc; else if (t == 1) acc1 = acc; else if (t == 2) acc2 = acc; else acc3 = acc;
    }
  }
  const int mbase = mT * 16 + ((lane & 16) ? 8 : 0);
  float* Cb = C + (long long)b * cStride;
#pragma unroll
  for (int t = 0; t < 4; ++t) {
    const v8f acc = (t == 0) ? acc0 : (t == 1) ? acc1 : (t == 2) ? acc2 : acc3;
    const int col = (nS * 4 + t) * 16 + (lane & 15);
    const float bv = bias ? bias[col] : 0.0f;
#pragma unroll
    for (int r = 0; r < 8; ++r) {
      int row = mbase + r;
      if (row < M) Cb[(long long)row * Nout + col] = acc[r] + bv;
    }
  }
}

// ---------------------------------------------------------------------------
// GCN aggregation
// ---------------------------------------------------------------------------

__global__ void k_agg_init(float* __restrict__ agg, const float* __restrict__ h,
                           const float* __restrict__ dinv, const float* __restrict__ bg,
                           long long total, int N) {
  long long i = (long long)blockIdx.x * blockDim.x + threadIdx.x;
  if (i >= total) return;
  int j = (int)(i & 127);
  long long nk = i >> 7;          // k*N + n
  float dv = dinv[nk];
  agg[i] = h[i] * dv * dv + bg[j];
}

__global__ void k_scatter(float* __restrict__ agg, const float* __restrict__ h,
                          const float* __restrict__ dinv, const int* __restrict__ edges,
                          long long E, int N) {
  const int wib  = threadIdx.x >> 5;
  const int lane = threadIdx.x & 31;
  const long long e = (long long)blockIdx.x * 8 + wib;
  if (e >= E) return;
  const int r = blockIdx.y;
  const int p = (r == 0) ? 3 : (r - 1);
  const int* eb = edges + (long long)p * 2 * E;
  const int src = eb[e], dst = eb[E + e];
  const float nrm = dinv[(long long)r * N + src] * dinv[(long long)r * N + dst];
  const float4 hv = *(const float4*)(h + (((long long)r * N + src) << 7) + lane * 4);
  float* ap = agg + (((long long)r * N + dst) << 7) + lane * 4;
  atomicAdd(ap + 0, hv.x * nrm);
  atomicAdd(ap + 1, hv.y * nrm);
  atomicAdd(ap + 2, hv.z * nrm);
  atomicAdd(ap + 3, hv.w * nrm);
}

// ---------------------------------------------------------------------------
// BatchNorm (training stats over N) + leaky_relu residual update
// ---------------------------------------------------------------------------

__global__ void k_bn_stats(float* __restrict__ stats, const float* __restrict__ agg, int N) {
  const int k = blockIdx.y, j = threadIdx.x;
  const int n0 = blockIdx.x * 256;
  const int n1 = min(n0 + 256, N);
  float s = 0.f, s2 = 0.f;
  for (int n = n0; n < n1; ++n) {
    float v = agg[(((long long)k * N + n) << 7) + j];
    s += v; s2 += v * v;
  }
  atomicAdd(&stats[(k * 128 + j) * 2 + 0], s);
  atomicAdd(&stats[(k * 128 + j) * 2 + 1], s2);
}

__global__ void k_bn_final(float* __restrict__ scsh, const float* __restrict__ stats,
                           const float* __restrict__ gamma, const float* __restrict__ beta,
                           int N) {
  int t = blockIdx.x * blockDim.x + threadIdx.x;
  if (t >= 512) return;
  int j = t & 127;
  float inv = 1.0f / (float)N;
  float mean = stats[t * 2 + 0] * inv;
  float var  = stats[t * 2 + 1] * inv - mean * mean;
  float sc = gamma[j] * rsqrtf(var + 1e-5f);
  scsh[t]       = sc;
  scsh[512 + t] = beta[j] - mean * sc;
}

__global__ void k_update(float* __restrict__ emb, const float* __restrict__ agg,
                         const float* __restrict__ scsh, long long total, int N) {
  long long i = (long long)blockIdx.x * blockDim.x + threadIdx.x;
  if (i >= total) return;
  int j = (int)(i & 127);
  int k = (int)(i / ((long long)N << 7));
  int kj = k * 128 + j;
  float v = agg[i] * scsh[kj] + scsh[512 + kj];
  emb[i] += lrelu(v);
}

__global__ void k_rels_mm(float* __restrict__ out, const float* __restrict__ rin,
                          const float* __restrict__ W, const float* __restrict__ b) {
  int t = blockIdx.x * blockDim.x + threadIdx.x;
  if (t >= 512) return;
  int k = t >> 7, o = t & 127;
  float a = b[o];
  for (int j = 0; j < 128; ++j) a += rin[k * 128 + j] * W[o * 128 + j];
  out[t] = a;
}

// ---------------------------------------------------------------------------
// Per-node 4x4 multi-head attention (L=4, heads=4, hd=32): one wave per head.
// qkv chunk layout [4][Mc][384]; attn_out global layout [4][N][128].
// ---------------------------------------------------------------------------

__global__ void k_attn(float* __restrict__ attn_out, const float* __restrict__ qkv,
                       int N, int Mc, int c0) {
  const int ln   = blockIdx.x;
  const int h    = threadIdx.x >> 5;
  const int lane = threadIdx.x & 31;
  const float qs = 0.17677669529663687f;  // 1/sqrt(32)
  float q[4], kk[4], vv[4];
#pragma unroll
  for (int l = 0; l < 4; ++l) {
    const float* p = qkv + ((long long)l * Mc + ln) * 384 + h * 32 + lane;
    q[l]  = p[0] * qs;
    kk[l] = p[128];
    vv[l] = p[256];
  }
  float s[4][4];
#pragma unroll
  for (int l = 0; l < 4; ++l)
#pragma unroll
    for (int m = 0; m < 4; ++m) {
      float p = q[l] * kk[m];
      for (int off = 16; off > 0; off >>= 1) p += __shfl_xor(p, off, 32);
      s[l][m] = p;
    }
#pragma unroll
  for (int l = 0; l < 4; ++l) {
    float mx = fmaxf(fmaxf(s[l][0], s[l][1]), fmaxf(s[l][2], s[l][3]));
    float e0 = expf(s[l][0] - mx), e1 = expf(s[l][1] - mx);
    float e2 = expf(s[l][2] - mx), e3 = expf(s[l][3] - mx);
    float inv = 1.0f / (e0 + e1 + e2 + e3);
    float o = (e0 * vv[0] + e1 * vv[1] + e2 * vv[2] + e3 * vv[3]) * inv;
    attn_out[(((long long)l * N + (c0 + ln)) << 7) + h * 32 + lane] = o;
  }
}

// f[l] = fusion[l]*alpha[l] + (1-alpha[l])*stk[l], stk[l] = emb[{3,0,2,1}[l]]
__global__ void k_blend(float* __restrict__ f, const float* __restrict__ fusion,
                        const float* __restrict__ emb, const float* __restrict__ alphas,
                        long long total, int N) {
  long long i = (long long)blockIdx.x * blockDim.x + threadIdx.x;
  if (i >= total) return;
  long long sl = (long long)N << 7;
  int l = (int)(i / sl);
  long long r = i % sl;
  int sm = (l == 0) ? 3 : (l == 1) ? 0 : (l == 2) ? 2 : 1;
  float a = alphas[l];
  f[i] = fusion[i] * a + (1.0f - a) * emb[(long long)sm * sl + r];
}

__global__ void k_score_reduce(float* __restrict__ acc, const float* __restrict__ y,
                               const float* __restrict__ fq, int N) {
  __shared__ float sm[128];
  const int l = blockIdx.y, j = threadIdx.x;
  const int n0 = blockIdx.x * 256;
  const int n1 = min(n0 + 256, N);
  const float qj = fq[j];
  float a = 0.f;
  for (int n = n0; n < n1; ++n)
    a += lrelu(y[(((long long)l * N + n) << 7) + j]) * qj;
  sm[j] = a;
  __syncthreads();
  for (int s2 = 64; s2 > 0; s2 >>= 1) {
    if (j < s2) sm[j] += sm[j + s2];
    __syncthreads();
  }
  if (j == 0) atomicAdd(&acc[l], sm[0]);
}

// softmax over [S0, S1, S3, S2]; region coefs c[l] for f[l]: c={w0,w1,w3,w2}
__global__ void k_weights(float* __restrict__ coef, const float* __restrict__ acc, int N) {
  if (threadIdx.x != 0 || blockIdx.x != 0) return;
  float inv = 1.0f / (float)N;
  float a0 = acc[0] * inv, a1 = acc[1] * inv, a2 = acc[3] * inv, a3 = acc[2] * inv;
  float mx = fmaxf(fmaxf(a0, a1), fmaxf(a2, a3));
  float e0 = expf(a0 - mx), e1 = expf(a1 - mx), e2 = expf(a2 - mx), e3 = expf(a3 - mx);
  float s = 1.0f / (e0 + e1 + e2 + e3);
  coef[0] = e0 * s;  // w0 -> f0
  coef[1] = e1 * s;  // w1 -> f1
  coef[2] = e3 * s;  // w3 -> f2 (d_f blended with w[3])
  coef[3] = e2 * s;  // w2 -> f3 (s_f blended with w[2])
}

__global__ void k_outputs(float* __restrict__ out, const float* __restrict__ f,
                          const float* __restrict__ coef, const float* __restrict__ rels,
                          long long slice) {
  long long i = (long long)blockIdx.x * blockDim.x + threadIdx.x;
  if (i >= slice) return;
  int j = (int)(i & 127);
  float reg = coef[0] * f[i] + coef[1] * f[slice + i] +
              coef[2] * f[2 * slice + i] + coef[3] * f[3 * slice + i];
  out[i]             = reg;
  out[slice + i]     = reg * rels[3 * 128 + j];
  out[2 * slice + i] = reg * rels[0 * 128 + j];
  out[3 * slice + i] = reg * rels[1 * 128 + j];
  out[4 * slice + i] = reg * rels[2 * 128 + j];
}

// ---------------------------------------------------------------------------
// Host orchestration
// ---------------------------------------------------------------------------

extern "C" void kernel_launch(void* const* d_in, const int* in_sizes, int n_in,
                              void* d_out, int out_size, void* d_ws, size_t ws_size,
                              hipStream_t stream) {
  const int d = 128;
  const int N = in_sizes[0] / d;
  const long long E = (long long)in_sizes[2] / 8;  // [4,2,E]

  const float* features = (const float*)d_in[0];
  const float* rel_emb  = (const float*)d_in[1];
  const int*   edges    = (const int*)d_in[2];
  const float* W_gcn    = (const float*)d_in[3];
  const float* b_gcn    = (const float*)d_in[4];
  const float* bn_gamma = (const float*)d_in[5];
  const float* bn_beta  = (const float*)d_in[6];
  const float* W_rt     = (const float*)d_in[7];
  const float* b_rt     = (const float*)d_in[8];
  const float* w_in     = (const float*)d_in[9];
  const float* b_in     = (const float*)d_in[10];
  const float* w_out    = (const float*)d_in[11];
  const float* b_out    = (const float*)d_in[12];
  const float* alphas   = (const float*)d_in[13];
  const float* fq       = (const float*)d_in[14];
  const float* fw       = (const float*)d_in[15];
  const float* fb       = (const float*)d_in[16];

  const long long EMBF  = 4LL * N * d;       // one [4,N,128] buffer, in floats
  const long long SLICE = (long long)N * d;

  float* ws      = (float*)d_ws;
  float* emb     = ws;                       // [4,N,128] embeddings (processing order n,poi,s,d)
  float* bufA    = emb + EMBF;               // h / attn_out / f
  float* bufB    = bufA + EMBF;              // agg / qkv-chunk / fusion / y
  float* dinv    = bufB + EMBF;              // [4,N], doubles as deg accumulator
  float* relsA   = dinv + 4LL * N;           // [4,128]
  float* relsB   = relsA + 512;
  float* stats   = relsB + 512;              // [4,128,2]
  float* scsh    = stats + 1024;             // scale[512] + shift[512]
  float* scoreAcc = scsh + 1024;             // [4] (+pad)
  float* coef    = scoreAcc + 8;             // [4] (+pad)
  __bf16* packHi = (__bf16*)(coef + 8);      // up to 4*128*128 (conv) or 128*384 halfs
  __bf16* packLo = packHi + 4LL * 128 * 128;

  const int TPB = 256;
  auto blocks = [](long long n, int t) { return (unsigned)((n + t - 1) / t); };

  auto launch_gemm = [&](const float* A, long long aStride, int m0, int m1, int m2, int m3,
                         int rowOff, const __bf16* bhi, const __bf16* blo, long long bStride,
                         const float* bias, float* Cp, long long cStride, int M, int Nout) {
    long long strips = (long long)((M + 15) / 16) * (Nout / 64);
    dim3 g(blocks(strips, 8), 4);
    k_gemm<<<g, TPB, 0, stream>>>(A, aStride, m0, m1, m2, m3, rowOff,
                                  bhi, blo, bStride, bias, Cp, cStride, M, Nout);
  };

  // ---- init: broadcast features, init rels, degrees -> dinv ----
  k_emb_init<<<blocks(EMBF, TPB), TPB, 0, stream>>>(emb, features, EMBF, SLICE);
  k_rels_init<<<2, TPB, 0, stream>>>(relsA, rel_emb);
  hipMemsetAsync(dinv, 0, sizeof(float) * 4 * N, stream);
  k_deg<<<blocks(4 * E, TPB), TPB, 0, stream>>>(dinv, edges, E, N);
  k_dinv<<<blocks(4LL * N, TPB), TPB, 0, stream>>>(dinv, 4LL * N);

  const float* relsCur = relsA;
  float* relsNext = relsB;

  // ---- 3 relation-GCN layers ----
  for (int i = 0; i < 3; ++i) {
    // pack per-relation B' = diag(rels_k) @ W_gcn[i] into bf16 hi/lo fragments
    k_pack<<<blocks(4LL * d * d, TPB), TPB, 0, stream>>>(
        packHi, packLo, W_gcn + (long long)i * d * d, relsCur, 0, d, 4LL * d * d);
    // h = emb @ B'  (WMMA)
    launch_gemm(emb, SLICE, 0, 1, 2, 3, 0, packHi, packLo, (long long)d * d,
                nullptr, bufA, SLICE, N, d);
    // agg = h*dinv^2 + bias, then scatter-add edge messages
    k_agg_init<<<blocks(EMBF, TPB), TPB, 0, stream>>>(bufB, bufA, dinv, b_gcn + i * d, EMBF, N);
    k_scatter<<<dim3(blocks(E, 8), 4), TPB, 0, stream>>>(bufB, bufA, dinv, edges, E, N);
    if (i < 2) {
      hipMemsetAsync(stats, 0, sizeof(float) * 1024, stream);
      k_bn_stats<<<dim3(blocks(N, 256), 4), 128, 0, stream>>>(stats, bufB, N);
      k_bn_final<<<2, TPB, 0, stream>>>(scsh, stats, bn_gamma + i * d, bn_beta + i * d, N);
      k_update<<<blocks(EMBF, TPB), TPB, 0, stream>>>(emb, bufB, scsh, EMBF, N);
    } else {
      hipMemcpyAsync(emb, bufB, sizeof(float) * EMBF, hipMemcpyDeviceToDevice, stream);
    }
    k_rels_mm<<<2, TPB, 0, stream>>>(relsNext, relsCur, W_rt + (long long)i * d * d, b_rt + i * d);
    const float* t = relsCur; relsCur = relsNext; relsNext = (float*)t;
  }

  // ---- MHA fusion over stk = emb[{3,0,2,1}] ----
  // QKV chunked through bufB (4*Mc*384 <= 4*N*128 since Mc ~ N/5)
  k_pack<<<blocks((long long)d * 384, TPB), TPB, 0, stream>>>(
      packHi, packLo, w_in, nullptr, 1, 384, (long long)d * 384);
  {
    int Nc = (N + 4) / 5;
    Nc = ((Nc + 15) / 16) * 16;
    for (int c0 = 0; c0 < N; c0 += Nc) {
      int Mc = (N - c0 < Nc) ? (N - c0) : Nc;
      launch_gemm(emb, SLICE, 3, 0, 2, 1, c0, packHi, packLo, 0,
                  b_in, bufB, (long long)Mc * 384, Mc, 384);
      k_attn<<<Mc, 128, 0, stream>>>(bufA, bufB, N, Mc, c0);
    }
  }
  // output projection: fusion = attn_out @ w_out^T + b_out
  k_pack<<<blocks((long long)d * d, TPB), TPB, 0, stream>>>(
      packHi, packLo, w_out, nullptr, 1, d, (long long)d * d);
  launch_gemm(bufA, SLICE, 0, 1, 2, 3, 0, packHi, packLo, 0, b_out, bufB, SLICE, N, d);
  // blend: f (bufA) from fusion (bufB) and stk (emb)
  k_blend<<<blocks(EMBF, TPB), TPB, 0, stream>>>(bufA, bufB, emb, alphas, EMBF, N);
  // scores: y = f @ fusion_w^T + fusion_b, then sum(lrelu(y)*q)/N per l
  k_pack<<<blocks((long long)d * d, TPB), TPB, 0, stream>>>(
      packHi, packLo, fw, nullptr, 1, d, (long long)d * d);
  launch_gemm(bufA, SLICE, 0, 1, 2, 3, 0, packHi, packLo, 0, fb, bufB, SLICE, N, d);
  hipMemsetAsync(scoreAcc, 0, sizeof(float) * 4, stream);
  k_score_reduce<<<dim3(blocks(N, 256), 4), 128, 0, stream>>>(scoreAcc, bufB, fq, N);
  k_weights<<<1, 32, 0, stream>>>(coef, scoreAcc, N);
  // region + the four rels-scaled outputs
  k_outputs<<<blocks(SLICE, TPB), TPB, 0, stream>>>((float*)d_out, bufA, coef, relsCur, SLICE);
}
